// PredictionBiLSTM_54924041781763
// MI455X (gfx1250) — compile-verified, matrix operand order checked by reading, not run on hardware
//
#include <hip/hip_runtime.h>
#include <hip/hip_bf16.h>

// BiLSTM single step, zero state ⇒ one GEMM [131072 x 904] x [904 x 600] (i,g,o
// gates, f dropped since c0=0) + lane-local epilogue.  bf16 WMMA with fp32
// accumulate; weights pre-packed to gate-major bf16 planes so one wave holds
// i/g/o for the same 16 units x 16 rows -> fused epilogue with zero extra HBM
// traffic.  HBM floor ~580MB @ 23.3 TB/s ~= 25us.

typedef __bf16 bf16_t;
typedef __attribute__((ext_vector_type(8)))  bf16_t v8bf;
typedef __attribute__((ext_vector_type(16))) bf16_t v16bf;
typedef __attribute__((ext_vector_type(8)))  float  v8f;
typedef __attribute__((ext_vector_type(4)))  float  f32x4;

static constexpr int IN_DIM  = 904;
static constexpr int K_PAD   = 928;        // 29 * 32
static constexpr int K_STEPS = 29;
static constexpr int HID     = 100;
static constexpr int UPAD    = 112;        // hidden padded to 7 tiles of 16
static constexpr int U2      = 224;        // both directions
static constexpr int OUTW    = 200;
static constexpr int NCELLS  = 131072;
static constexpr int MBLK    = 64;         // rows per workgroup
static constexpr int ASTRIDE = 40;         // bf16 elems per LDS row (80B, bank-spread)

// ---------------- prep: pack weights to bf16 gate-major planes ---------------
// Wp layout: [q in {i,g,o}][u in 0..223][k in 0..927]; u<112 = fwd, else bwd;
// padded units / padded K are zero so the GEMM tail needs no guards.
__global__ void bilstm_pack_w(const float* __restrict__ wf,
                              const float* __restrict__ wb,
                              bf16_t* __restrict__ Wp) {
  int idx = blockIdx.x * 256 + threadIdx.x;
  if (idx >= 3 * U2 * K_PAD) return;
  int k = idx % K_PAD;
  int t = idx / K_PAD;
  int u = t % U2;
  int q = t / U2;
  int dir = u / UPAD, lu = u % UPAD;
  int goff = (q == 0) ? 0 : (q == 1 ? 2 * HID : 3 * HID);   // i, g, o rows of W
  float v = 0.0f;
  if (lu < HID && k < IN_DIM) {
    const float* w = dir ? wb : wf;
    v = w[(goff + lu) * IN_DIM + k];
  }
  Wp[idx] = (bf16_t)v;
}

// biasP layout: [q][u] fp32, b_ih + b_hh.
__global__ void bilstm_pack_b(const float* __restrict__ bif, const float* __restrict__ bhf,
                              const float* __restrict__ bib, const float* __restrict__ bhb,
                              float* __restrict__ biasP) {
  int idx = blockIdx.x * 256 + threadIdx.x;
  if (idx >= 3 * U2) return;
  int u = idx % U2, q = idx / U2;
  int dir = u / UPAD, lu = u % UPAD;
  int goff = (q == 0) ? 0 : (q == 1 ? 2 * HID : 3 * HID);
  float v = 0.0f;
  if (lu < HID)
    v = dir ? (bib[goff + lu] + bhb[goff + lu]) : (bif[goff + lu] + bhf[goff + lu]);
  biasP[idx] = v;
}

// ------------------------------- main kernel --------------------------------
__device__ __forceinline__ float sig_(float x) { return 1.0f / (1.0f + __expf(-x)); }

__global__ __launch_bounds__(256)
void bilstm_wmma_kernel(const float* __restrict__ X,
                        const bf16_t* __restrict__ Wp,
                        const float* __restrict__ biasP,
                        float* __restrict__ out) {
  // 8 waves: 4 row-subtiles x 2 unit-tiles.  grid = (7 tile-pairs, 2048 rowblocks)
  __shared__ __align__(16) bf16_t lds[2][MBLK * ASTRIDE];

  const int tid  = threadIdx.x;
  const int wave = tid >> 5;
  const int lane = tid & 31;
  const int half = lane >> 4;      // K-half selector in A/B fragment layout
  const int ln16 = lane & 15;
  const int mw   = wave & 3;                       // row sub-tile 0..3
  const int tw   = 2 * blockIdx.x + (wave >> 2);   // unit tile 0..13
  const long rowBase = (long)blockIdx.y * MBLK;

  // cooperative A staging role: 4 threads per row, 8 floats each
  const int srow = tid >> 2;
  const int schk = tid & 3;
  const float* srcRow = X + (rowBase + srow) * (long)IN_DIM + schk * 8;

  v8f acc0 = {}, acc1 = {}, acc2 = {};

  // prologue: stage K-step 0 (always fully in-bounds: k < 32 < 904)
  {
    f32x4 f0 = *(const f32x4*)(srcRow);
    f32x4 f1 = *(const f32x4*)(srcRow + 4);
    v8bf h;
    h[0] = (bf16_t)f0.x; h[1] = (bf16_t)f0.y; h[2] = (bf16_t)f0.z; h[3] = (bf16_t)f0.w;
    h[4] = (bf16_t)f1.x; h[5] = (bf16_t)f1.y; h[6] = (bf16_t)f1.z; h[7] = (bf16_t)f1.w;
    *(v8bf*)&lds[0][srow * ASTRIDE + schk * 8] = h;
  }
  __syncthreads();

  for (int kk = 0; kk < K_STEPS; ++kk) {
    const int cur = kk & 1;

    // ---- issue global loads for next K-step (double buffer) ----
    f32x4 f0 = {}, f1 = {};
    const int nk = kk + 1;
    const bool have = nk < K_STEPS;
    if (have) {
      int kb = nk * 32 + schk * 8;                 // chunk fully valid or fully pad
      if (kb + 8 <= IN_DIM) {
        f0 = *(const f32x4*)(srcRow + nk * 32);
        f1 = *(const f32x4*)(srcRow + nk * 32 + 4);
      }
    }
    if (kk + 2 < K_STEPS)
      __builtin_prefetch(srcRow + (kk + 2) * 32, 0, 1);   // global_prefetch_b8

    // ---- A fragment from LDS (16x32 bf16, ISA A layout) ----
    const bf16_t* ap = &lds[cur][(mw * 16 + ln16) * ASTRIDE + half * 8];
    v8bf alo = *(const v8bf*)ap;
    v8bf ahi = *(const v8bf*)(ap + 16);
    v16bf afrag = __builtin_shufflevector(alo, ahi,
        0, 1, 2, 3, 4, 5, 6, 7, 8, 9, 10, 11, 12, 13, 14, 15);

    // ---- B fragments: B^T rows in A addressing == valid B layout ----
    const bf16_t* bbase = Wp + ((long)(tw * 16 + ln16)) * K_PAD + kk * 32 + half * 8;
    {
      v8bf blo = *(const v8bf*)bbase;
      v8bf bhi = *(const v8bf*)(bbase + 16);
      v16bf bfrag = __builtin_shufflevector(blo, bhi,
          0, 1, 2, 3, 4, 5, 6, 7, 8, 9, 10, 11, 12, 13, 14, 15);
      acc0 = __builtin_amdgcn_wmma_f32_16x16x32_bf16(false, afrag, false, bfrag,
                                                     (short)0, acc0, false, false);
    }
    {
      const bf16_t* bp = bbase + (long)U2 * K_PAD;
      v8bf blo = *(const v8bf*)bp;
      v8bf bhi = *(const v8bf*)(bp + 16);
      v16bf bfrag = __builtin_shufflevector(blo, bhi,
          0, 1, 2, 3, 4, 5, 6, 7, 8, 9, 10, 11, 12, 13, 14, 15);
      acc1 = __builtin_amdgcn_wmma_f32_16x16x32_bf16(false, afrag, false, bfrag,
                                                     (short)0, acc1, false, false);
    }
    {
      const bf16_t* bp = bbase + 2L * U2 * K_PAD;
      v8bf blo = *(const v8bf*)bp;
      v8bf bhi = *(const v8bf*)(bp + 16);
      v16bf bfrag = __builtin_shufflevector(blo, bhi,
          0, 1, 2, 3, 4, 5, 6, 7, 8, 9, 10, 11, 12, 13, 14, 15);
      acc2 = __builtin_amdgcn_wmma_f32_16x16x32_bf16(false, afrag, false, bfrag,
                                                     (short)0, acc2, false, false);
    }

    // ---- store next buffer ----
    if (have) {
      v8bf h;
      h[0] = (bf16_t)f0.x; h[1] = (bf16_t)f0.y; h[2] = (bf16_t)f0.z; h[3] = (bf16_t)f0.w;
      h[4] = (bf16_t)f1.x; h[5] = (bf16_t)f1.y; h[6] = (bf16_t)f1.z; h[7] = (bf16_t)f1.w;
      *(v8bf*)&lds[cur ^ 1][srow * ASTRIDE + schk * 8] = h;
    }
    __syncthreads();
  }

  // ---- fused epilogue: lane-local across the three gate accumulators ----
  // C layout: lane -> column n = lane%16 (one hidden unit), VGPR r -> row r + 8*(lane/16)
  const int uu  = tw * 16 + ln16;
  const int dir = uu / UPAD;
  const int lu  = uu % UPAD;
  if (lu < HID) {
    const float bi = biasP[uu];
    const float bg = biasP[U2 + uu];
    const float bo = biasP[2 * U2 + uu];
    #pragma unroll
    for (int r = 0; r < 8; ++r) {
      long m = rowBase + mw * 16 + half * 8 + r;
      float gi = sig_(acc0[r] + bi);        // input gate
      float gg = tanhf(acc1[r] + bg);       // cell candidate
      float go = sig_(acc2[r] + bo);        // output gate
      out[m * OUTW + dir * HID + lu] = go * tanhf(gi * gg);
    }
  }
}

// ------------------------------- launcher -----------------------------------
extern "C" void kernel_launch(void* const* d_in, const int* in_sizes, int n_in,
                              void* d_out, int out_size, void* d_ws, size_t ws_size,
                              hipStream_t stream) {
  const float* V   = (const float*)d_in[0];
  const float* wf  = (const float*)d_in[1];
  // d_in[2] = w_hh_f (unused: zero state)
  const float* bif = (const float*)d_in[3];
  const float* bhf = (const float*)d_in[4];
  const float* wb  = (const float*)d_in[5];
  // d_in[6] = w_hh_b (unused)
  const float* bib = (const float*)d_in[7];
  const float* bhb = (const float*)d_in[8];
  float* out = (float*)d_out;

  float*  biasP = (float*)d_ws;                       // 3*224 fp32
  bf16_t* Wp    = (bf16_t*)((char*)d_ws + 4096);      // 3*224*928 bf16 ~ 1.25MB

  const int wtotal = 3 * U2 * K_PAD;
  bilstm_pack_w<<<(wtotal + 255) / 256, 256, 0, stream>>>(wf, wb, Wp);
  bilstm_pack_b<<<(3 * U2 + 255) / 256, 256, 0, stream>>>(bif, bhf, bib, bhb, biasP);

  dim3 grid(7, NCELLS / MBLK);   // unit-tile pairs x row blocks
  bilstm_wmma_kernel<<<grid, 256, 0, stream>>>(V, Wp, biasP, out);
}
